// SelfAttention_19473381720409
// MI455X (gfx1250) — matrix operand order repaired and checked by reading, not compiled, not run on hardware
//
#include <hip/hip_runtime.h>

typedef __attribute__((ext_vector_type(16))) _Float16 v16h;
typedef __attribute__((ext_vector_type(8)))  _Float16 v8h;
typedef __attribute__((ext_vector_type(8)))  float    v8f;

#define S_LEN 4096
#define NHEAD 16
#define HDIM  64
#define BM    128   // query rows per workgroup (8 waves x 16 rows)
#define BN    64    // key rows per iteration (two K=32 WMMA steps for PV)
#define NWAVE 8
#define KT_S  72    // padded row stride (halves) for transposed K tile [dim][key]
#define V_S   72    // padded row stride (halves) for V tile [key][dim]
#define P_S   72    // padded row stride (halves) for P staging [row][key]

static __device__ __forceinline__ v16h join8(v8h lo, v8h hi) {
  return __builtin_shufflevector(lo, hi, 0,1,2,3,4,5,6,7,8,9,10,11,12,13,14,15);
}

// raw v_exp_f32 (flush-to-zero on subnormal output -- fine for softmax weights)
static __device__ __forceinline__ float fast_exp2(float x) {
  return __builtin_amdgcn_exp2f(x);
}

// DPP16 lane permute (stays within each 16-lane row -> matches C-layout groups)
template <int CTRL>
static __device__ __forceinline__ float permf(float x) {
  return __builtin_bit_cast(float, __builtin_amdgcn_update_dpp(
      0, __builtin_bit_cast(int, x), CTRL, 0xf, 0xf, true));
}
// butterfly reduce-to-all over 16 lanes: quad xor1, quad xor2, half-mirror, mirror
static __device__ __forceinline__ float rowmax16(float x) {
  x = fmaxf(x, permf<0xB1>(x));    // quad_perm [1,0,3,2]
  x = fmaxf(x, permf<0x4E>(x));    // quad_perm [2,3,0,1]
  x = fmaxf(x, permf<0x141>(x));   // row_half_mirror
  x = fmaxf(x, permf<0x140>(x));   // row_mirror
  return x;
}
static __device__ __forceinline__ float rowsum16(float x) {
  x += permf<0xB1>(x);
  x += permf<0x4E>(x);
  x += permf<0x141>(x);
  x += permf<0x140>(x);
  return x;
}

__global__ __launch_bounds__(256)
void fattn_fwd_19473381720409(const float* __restrict__ qkv, float* __restrict__ out) {
  const int qblk = blockIdx.x;
  const int head = blockIdx.y;
  const int tid  = threadIdx.x;
  const int wave = tid >> 5;
  const int lane = tid & 31;
  const int ln16 = lane & 15;
  const int hi   = lane >> 4;

  __shared__ _Float16 kT[HDIM * KT_S];          // K tile, transposed: [dim][key]
  __shared__ _Float16 vT[BN * V_S];             // V tile, row-major:  [key][dim]
  __shared__ _Float16 pS[NWAVE][16 * P_S];      // per-wave P staging  [row][key]

  const int qbase = qblk * BM;
  const int qrow0 = qbase + wave * 16;          // first query row of this wave
  const int qEnd  = qrow0 + 15;

  const float sscale = 0.125f * 1.44269504088896340736f;  // 1/sqrt(64) * log2(e)

  // ---- Q A-fragments, softmax scale folded in (fp32 -> f16), two K=32 chunks
  v16h qf[2];
  {
    const float* qp = qkv + (size_t)(qrow0 + ln16) * 3072 + head * 64; // t=0 (Q)
    #pragma unroll
    for (int c = 0; c < 2; ++c) {
      v16h q;
      #pragma unroll
      for (int e = 0; e < 8; ++e) q[e]     = (_Float16)(qp[32*c + 8*hi + e] * sscale);
      #pragma unroll
      for (int e = 0; e < 8; ++e) q[8 + e] = (_Float16)(qp[32*c + 16 + 8*hi + e] * sscale);
      qf[c] = q;
    }
  }

  v8f   ofrag[4] = {};                          // O accumulator, 4 x 16 columns
  float m8[8], l8[8];
  #pragma unroll
  for (int r = 0; r < 8; ++r) { m8[r] = -__builtin_inff(); l8[r] = 0.f; }

  const int numKB = (qbase + BM) / BN;
  for (int kb = 0; kb < numKB; ++kb) {
    const int k0 = kb * BN;
    __syncthreads();                            // protect tiles from previous iter readers
    // cooperative K/V tile load + fp32->f16 convert (coalesced b32 reads)
    for (int i = tid; i < BN * HDIM; i += 256) {
      const int key = i >> 6, d = i & 63;
      const size_t off = (size_t)(k0 + key) * 3072 + head * 64 + d;
      kT[d * KT_S + key] = (_Float16)qkv[off + 1024];   // t=1 (K), transposed
      vT[key * V_S + d]  = (_Float16)qkv[off + 2048];   // t=2 (V)
    }
    // prefetch next tile's cachelines: 2 tensors x 64 keys x 4 lines = 512
    if (kb + 1 < numKB) {
      for (int t = tid; t < 512; t += 256) {
        const int key = (t >> 2) & 63, tens = 1 + (t >> 8), line = t & 3;
        __builtin_prefetch(qkv + (size_t)(k0 + BN + key) * 3072 + tens * 1024 +
                           head * 64 + line * 16, 0, 0);
      }
    }
    __syncthreads();

    if (k0 > qEnd) continue;                    // this wave fully masked for this block

    // ---- S = Q * K^T : four 16-column chunks, accumulate over D in two K=32 steps
    v8f sf[4] = {};
    #pragma unroll
    for (int j = 0; j < 4; ++j) {
      #pragma unroll
      for (int c = 0; c < 2; ++c) {
        const _Float16* bp = &kT[(32 * c + 8 * hi) * KT_S + 16 * j + ln16];
        v16h bk;
        #pragma unroll
        for (int e = 0; e < 8; ++e) { bk[e] = bp[e * KT_S]; bk[8 + e] = bp[(16 + e) * KT_S]; }
        sf[j] = __builtin_amdgcn_wmma_f32_16x16x32_f16(false, qf[c], false, bk,
                                                       (short)0, sf[j], false, false);
      }
    }

    // ---- causal mask only on diagonal blocks (wave-uniform branch)
    const int rowg = qrow0 + 8 * hi;
    if (k0 + BN - 1 > qrow0) {
      #pragma unroll
      for (int j = 0; j < 4; ++j) {
        const int col = k0 + 16 * j + ln16;
        #pragma unroll
        for (int r = 0; r < 8; ++r)
          if (col > rowg + r) sf[j][r] = -__builtin_inff();
      }
    }

    // ---- online softmax (rows r + 8*hi); DPP butterfly reductions, no DS traffic
    _Float16* pw = &pS[wave][0];
    #pragma unroll
    for (int r = 0; r < 8; ++r) {
      float mx = fmaxf(fmaxf(sf[0][r], sf[1][r]), fmaxf(sf[2][r], sf[3][r]));
      mx = rowmax16(mx);
      const float mnew  = fmaxf(m8[r], mx);
      const float alpha = fast_exp2(m8[r] - mnew);  // first block: exp2(-inf)=0
      m8[r] = mnew;
      float rs = 0.f;
      #pragma unroll
      for (int j = 0; j < 4; ++j) {
        const float p = fast_exp2(sf[j][r] - mnew);
        rs += p;
        pw[(r + 8 * hi) * P_S + 16 * j + ln16] = (_Float16)p;
      }
      rs = rowsum16(rs);
      l8[r] = l8[r] * alpha + rs;
      #pragma unroll
      for (int j4 = 0; j4 < 4; ++j4) ofrag[j4][r] *= alpha;
    }

    // ---- read P back as A-fragments (same-wave DS ops are in-order)
    v16h pa[2];
    #pragma unroll
    for (int kk = 0; kk < 2; ++kk) {
      const _Float16* pp = &pw[ln16 * P_S + 32 * kk + 8 * hi];
      pa[kk] = join8(*(const v8h*)pp, *(const v8h*)(pp + 16));
    }

    // ---- O += P * V : two K=32 WMMA steps per 16-column output chunk
    #pragma unroll
    for (int j4 = 0; j4 < 4; ++j4) {
      #pragma unroll
      for (int kk = 0; kk < 2; ++kk) {
        const _Float16* vp = &vT[(32 * kk + 8 * hi) * V_S + 16 * j4 + ln16];
        v16h bv;
        #pragma unroll
        for (int e = 0; e < 8; ++e) { bv[e] = vp[e * V_S]; bv[8 + e] = vp[(16 + e) * V_S]; }
        ofrag[j4] = __builtin_amdgcn_wmma_f32_16x16x32_f16(false, pa[kk], false, bv,
                                                           (short)0, ofrag[j4], false, false);
      }
    }
  }

  // ---- normalize by softmax denominator and store (out is [1, S, H, D] fp32)
  #pragma unroll
  for (int r = 0; r < 8; ++r) {
    const float inv = __builtin_amdgcn_rcpf(l8[r]);
    float* op = out + (size_t)(qrow0 + r + 8 * hi) * (NHEAD * HDIM) + head * HDIM + ln16;
    #pragma unroll
    for (int j4 = 0; j4 < 4; ++j4)
      op[16 * j4] = ofrag[j4][r] * inv;
  }
}

extern "C" void kernel_launch(void* const* d_in, const int* in_sizes, int n_in,
                              void* d_out, int out_size, void* d_ws, size_t ws_size,
                              hipStream_t stream) {
  (void)in_sizes; (void)n_in; (void)out_size; (void)d_ws; (void)ws_size;
  const float* qkv = (const float*)d_in[0];
  float*       out = (float*)d_out;
  dim3 grid(S_LEN / BM, NHEAD);
  fattn_fwd_19473381720409<<<grid, 256, 0, stream>>>(qkv, out);
}
